// h_s_38972533243996
// MI455X (gfx1250) — 8-bit matrix operand order repaired and checked by reading, compiled, not run on hardware
//
#include <hip/hip_runtime.h>
#include <hip/hip_bf16.h>
#include <stdint.h>


typedef int      v8i __attribute__((ext_vector_type(8)));
typedef uint32_t v4u __attribute__((ext_vector_type(4)));

#define KCLP  10
#define SCL0K 4

__device__ __forceinline__ v8i load_a_frag(const int8_t* wb, int h) {
  const uint64_t q0 = *(const uint64_t*)(wb +  0 + (h << 4));
  const uint64_t q1 = *(const uint64_t*)(wb +  8 + (h << 4));
  const uint64_t q2 = *(const uint64_t*)(wb + 32 + (h << 4));
  const uint64_t q3 = *(const uint64_t*)(wb + 40 + (h << 4));
  v8i a;
  a[0] = (int)(uint32_t)q0; a[1] = (int)(uint32_t)(q0 >> 32);
  a[2] = (int)(uint32_t)q1; a[3] = (int)(uint32_t)(q1 >> 32);
  a[4] = (int)(uint32_t)q2; a[5] = (int)(uint32_t)(q2 >> 32);
  a[6] = (int)(uint32_t)q3; a[7] = (int)(uint32_t)(q3 >> 32);
  return a;
}

__device__ __forceinline__ v8i load_b_frag(const uint8_t* ab, bool inb) {
  v4u lo = {0u, 0u, 0u, 0u};
  v4u hi = {0u, 0u, 0u, 0u};
  if (inb) {
    lo = *(const v4u*)(ab);
    hi = *(const v4u*)(ab + 32);
  }
  v8i b;
  b[0] = (int)lo.x; b[1] = (int)lo.y; b[2] = (int)lo.z; b[3] = (int)lo.w;
  b[4] = (int)hi.x; b[5] = (int)hi.y; b[6] = (int)hi.z; b[7] = (int)hi.w;
  return b;
}

__device__ __forceinline__ uint32_t requant_one(int acc, int bias_i, int mul,
                                                int md, long long clp, long long scl) {
  long long v = (long long)(acc + bias_i) * (long long)mul;
  long long t = (v + (1LL << (md - 1 - KCLP))) >> (md - KCLP);
  t = t < 0 ? 0 : (t > clp ? clp : t);
  t = (t * scl + (1LL << (15 + KCLP - SCL0K))) >> (16 + KCLP - SCL0K);
  return (uint32_t)(t > 255 ? 255 : (t < 0 ? 0 : t));
}

__global__ __launch_bounds__(256) void deconv_iu8_wmma(
    const uint8_t* __restrict__ in,
    const int8_t*  __restrict__ wpk,
    const float*   __restrict__ bias,
    const float*   __restrict__ muls,
    const int* __restrict__ mdP, const int* __restrict__ reluP,
    uint8_t* __restrict__ out,
    int Bn, int Cin, int Cout, int CoutStride, int Hin, int Win) {
  const int Hout = Hin * 2, Wout = Win * 2;
  const int nseg  = Wout >> 5;
  const int Mblk2 = Cout >> 5;
  const int lane = threadIdx.x & 31;
  int tile = blockIdx.x * (blockDim.x >> 5) + (threadIdx.x >> 5);
  if (tile >= Mblk2 * Bn * Hout * 2 * nseg) return;
  const int mb2 = tile % Mblk2; tile /= Mblk2;
  const int seg = tile % nseg;  tile /= nseg;
  const int par = tile & 1;     tile >>= 1;
  const int oy  = tile % Hout;
  const int n   = tile / Hout;

  const int col = lane & 15;
  const int h   = lane >> 4;
  const int row = lane & 15;
  const int kblocks = Cin >> 6;
  const int co0 = mb2 << 5;

  v8i acc0 = {0,0,0,0,0,0,0,0};
  v8i acc1 = {0,0,0,0,0,0,0,0};

  for (int ky = (oy & 1); ky < 5; ky += 2) {
    const int iy = (oy + 2 - ky) >> 1;
    if (iy < 0 || iy >= Hin) continue;
    for (int kx = par; kx < 5; kx += 2) {
      const int ixc  = ((par + 2 - kx) >> 1) + (seg << 4) + col;
      const bool inb = (ixc >= 0) && (ixc < Win);
      const int  ixs = inb ? ixc : 0;
      const uint8_t* abase = in + ((size_t)((n * Hin + iy) * Win + ixs)) * Cin + (h << 4);
      const int8_t*  wA0   = wpk + (((size_t)(co0 + row) * 5 + ky) * 5 + kx) * Cin;
      const int8_t*  wA1   = wA0 + (size_t)16 * 25 * Cin;
      for (int kb = 0; kb < kblocks; ++kb) {
        v8i bfr = load_b_frag(abase + (kb << 6), inb);
        v8i af0 = load_a_frag(wA0 + (kb << 6), h);
        v8i af1 = load_a_frag(wA1 + (kb << 6), h);
        acc0 = __builtin_amdgcn_wmma_i32_16x16x64_iu8(true, af0, false, bfr, acc0, false, false);
        acc1 = __builtin_amdgcn_wmma_i32_16x16x64_iu8(true, af1, false, bfr, acc1, false, false);
      }
    }
  }

  const int md   = *mdP;
  const int relu = *reluP;
  const long long clp = (long long)(255.0 * 67108864.0 / (double)relu + 0.5);
  const long long scl = (long long)((relu + (1 << (SCL0K - 1))) >> SCL0K);
  const int ox = par + (((seg << 4) + col) << 1);
  uint8_t* obase = out + ((size_t)((n * Hout + oy) * Wout + ox)) * CoutStride + co0 + (h << 3);
  uint64_t p0 = 0, p1 = 0;
  for (int r = 0; r < 8; ++r) {
    const int co = co0 + (h << 3) + r;
    const uint32_t u0 = requant_one(acc0[r], __float2int_rn(bias[co]),
                                    __float2int_rn(muls[co]), md, clp, scl);
    const uint32_t u1 = requant_one(acc1[r], __float2int_rn(bias[co + 16]),
                                    __float2int_rn(muls[co + 16]), md, clp, scl);
    p0 |= (uint64_t)u0 << (r << 3);
    p1 |= (uint64_t)u1 << (r << 3);
  }
  *(uint64_t*)(obase)      = p0;
  *(uint64_t*)(obase + 16) = p1;
}

__global__ __launch_bounds__(256) void conv_iu8_wmma(
    const uint8_t* __restrict__ in,
    const int8_t*  __restrict__ wpk,
    const float* __restrict__ bias, const float* __restrict__ muls,
    const int* __restrict__ mdP, const int* __restrict__ hslP,
    float* __restrict__ out,
    int Bn, int Cinp, int Cout, int H, int W) {
  const int nseg  = W >> 4;
  const int Mblk2 = Cout >> 5;
  const int lane = threadIdx.x & 31;
  int tile = blockIdx.x * (blockDim.x >> 5) + (threadIdx.x >> 5);
  if (tile >= Mblk2 * Bn * H * nseg) return;
  const int mb2 = tile % Mblk2; tile /= Mblk2;
  const int seg = tile % nseg;  tile /= nseg;
  const int oy  = tile % H;
  const int n   = tile / H;

  const int col = lane & 15, h = lane >> 4, row = lane & 15;
  const int kblocks = Cinp >> 6;
  const int co0 = mb2 << 5;
  v8i acc0 = {0,0,0,0,0,0,0,0};
  v8i acc1 = {0,0,0,0,0,0,0,0};

  for (int ky = 0; ky < 3; ++ky) {
    const int iy = oy - 1 + ky;
    if (iy < 0 || iy >= H) continue;
    for (int kx = 0; kx < 3; ++kx) {
      const int ixc  = (seg << 4) + col - 1 + kx;
      const bool inb = (ixc >= 0) && (ixc < W);
      const int  ixs = inb ? ixc : 0;
      const uint8_t* abase = in + ((size_t)((n * H + iy) * W + ixs)) * Cinp + (h << 4);
      const int8_t*  wA0   = wpk + (((size_t)(co0 + row) * 3 + ky) * 3 + kx) * Cinp;
      const int8_t*  wA1   = wA0 + (size_t)16 * 9 * Cinp;
      for (int kb = 0; kb < kblocks; ++kb) {
        v8i bfr = load_b_frag(abase + (kb << 6), inb);
        v8i af0 = load_a_frag(wA0 + (kb << 6), h);
        v8i af1 = load_a_frag(wA1 + (kb << 6), h);
        acc0 = __builtin_amdgcn_wmma_i32_16x16x64_iu8(true, af0, false, bfr, acc0, false, false);
        acc1 = __builtin_amdgcn_wmma_i32_16x16x64_iu8(true, af1, false, bfr, acc1, false, false);
      }
    }
  }

  const int md = *mdP, hsl = *hslP;
  const int sh = md - hsl;
  const long long add = 1LL << (sh - 1);
  const int ox = (seg << 4) + col;
  for (int r = 0; r < 8; ++r) {
    const int co = co0 + (h << 3) + r;
    const long long v0 = (long long)(acc0[r] + __float2int_rn(bias[co]))
                         * (long long)__float2int_rn(muls[co]);
    const long long v1 = (long long)(acc1[r] + __float2int_rn(bias[co + 16]))
                         * (long long)__float2int_rn(muls[co + 16]);
    out[(((size_t)n * Cout + co)      * H + oy) * W + ox] = (float)((v0 + add) >> sh);
    out[(((size_t)n * Cout + co + 16) * H + oy) * W + ox] = (float)((v1 + add) >> sh);
  }
}

__global__ void pack_w_deconv(const float* __restrict__ w, int8_t* __restrict__ wpk,
                              int Cin, int Cout, int KH, int KW) {
  int idx = blockIdx.x * blockDim.x + threadIdx.x;
  const int total = Cout * KH * KW * Cin;
  if (idx >= total) return;
  const int ci = idx % Cin; int t = idx / Cin;
  const int kx = t % KW; t /= KW;
  const int ky = t % KH; t /= KH;
  const int co = t;
  const float v = w[(((size_t)ci * Cout + co) * KH + ky) * KW + kx];
  wpk[idx] = (int8_t)__float2int_rn(v);
}

__global__ void pack_w_conv(const float* __restrict__ w, int8_t* __restrict__ wpk,
                            int Cin, int Cinp, int Cout, int KH, int KW) {
  int idx = blockIdx.x * blockDim.x + threadIdx.x;
  const int total = Cout * KH * KW * Cinp;
  if (idx >= total) return;
  const int ci = idx % Cinp; int t = idx / Cinp;
  const int kx = t % KW; t /= KW;
  const int ky = t % KH; t /= KH;
  const int co = t;
  int8_t v = 0;
  if (ci < Cin)
    v = (int8_t)__float2int_rn(w[(((size_t)co * Cin + ci) * KH + ky) * KW + kx]);
  wpk[idx] = v;
}

__global__ void cvt_x_u8(const float* __restrict__ x, uint8_t* __restrict__ xu,
                         int Bn, int C, int H, int W) {
  int idx = blockIdx.x * blockDim.x + threadIdx.x;
  const int total = Bn * C * H * W;
  if (idx >= total) return;
  const int c = idx % C; int t = idx / C;
  const int xx = t % W; t /= W;
  const int y = t % H;
  const int n = t / H;
  xu[idx] = (uint8_t)__float2int_rn(x[(((size_t)n * C + c) * H + y) * W + xx]);
}

__global__ void zero_pad_tail(uint8_t* __restrict__ a, int npix, int Cstride, int Cvalid) {
  int idx = blockIdx.x * blockDim.x + threadIdx.x;
  if (idx >= npix) return;
  uint64_t* p = (uint64_t*)(a + (size_t)idx * Cstride + Cvalid);
  p[0] = 0; p[1] = 0; p[2] = 0; p[3] = 0;
}

extern "C" void kernel_launch(void* const* d_in, const int* in_sizes, int n_in,
                              void* d_out, int out_size, void* d_ws, size_t ws_size,
                              hipStream_t stream) {
  const float* x  = (const float*)d_in[0];
  const float* w0 = (const float*)d_in[1];
  const float* b0 = (const float*)d_in[2];
  const float* w1 = (const float*)d_in[3];
  const float* b1 = (const float*)d_in[4];
  const float* w2 = (const float*)d_in[5];
  const float* b2 = (const float*)d_in[6];
  const float* m0 = (const float*)d_in[7];
  const float* m1 = (const float*)d_in[8];
  const float* m2 = (const float*)d_in[9];
  const int* md0   = (const int*)d_in[10];
  const int* md1   = (const int*)d_in[11];
  const int* md2   = (const int*)d_in[12];
  const int* relu0 = (const int*)d_in[13];
  const int* relu1 = (const int*)d_in[14];
  const int* hsl   = (const int*)d_in[15];
  float* out = (float*)d_out;

  char* ws = (char*)d_ws;
  auto au = [](size_t v) { return (v + 255) & ~(size_t)255; };
  size_t o = 0;
  uint8_t* xu8 = (uint8_t*)(ws + o); o = au(o + (size_t)16 * 16 * 16 * 192);
  uint8_t* h0  = (uint8_t*)(ws + o); o = au(o + (size_t)16 * 32 * 32 * 320);
  uint8_t* h1  = (uint8_t*)(ws + o); o = au(o + (size_t)16 * 64 * 64 * 512);
  int8_t*  wp0 = (int8_t*)(ws + o);  o = au(o + (size_t)320 * 25 * 192);
  int8_t*  wp1 = (int8_t*)(ws + o);  o = au(o + (size_t)480 * 25 * 320);
  int8_t*  wp2 = (int8_t*)(ws + o);  o = au(o + (size_t)640 * 9 * 512);

  { const int t = 16 * 192 * 16 * 16;
    cvt_x_u8<<<(t + 255) / 256, 256, 0, stream>>>(x, xu8, 16, 192, 16, 16); }
  { const int t = 320 * 25 * 192;
    pack_w_deconv<<<(t + 255) / 256, 256, 0, stream>>>(w0, wp0, 192, 320, 5, 5); }
  { const int t = 480 * 25 * 320;
    pack_w_deconv<<<(t + 255) / 256, 256, 0, stream>>>(w1, wp1, 320, 480, 5, 5); }
  { const int t = 640 * 9 * 512;
    pack_w_conv<<<(t + 255) / 256, 256, 0, stream>>>(w2, wp2, 480, 512, 640, 3, 3); }
  { const int t = 16 * 64 * 64;
    zero_pad_tail<<<(t + 255) / 256, 256, 0, stream>>>(h1, t, 512, 480); }

  deconv_iu8_wmma<<<1280, 256, 0, stream>>>(xu8, wp0, b0, m0, md0, relu0,
                                            h0, 16, 192, 320, 320, 16, 16);
  deconv_iu8_wmma<<<7680, 256, 0, stream>>>(h0, wp1, b1, m1, md1, relu1,
                                            h1, 16, 320, 480, 512, 32, 32);
  conv_iu8_wmma<<<10240, 256, 0, stream>>>(h1, wp2, b2, m2, md2, hsl,
                                           out, 16, 512, 640, 64, 64);
}
